// ResiLevelTensorProductScoreModel_40200893891320
// MI455X (gfx1250) — matrix operand order repaired and checked by reading, not compiled, not run on hardware
//
#include <hip/hip_runtime.h>

#define N_NODES 10000
#define N_EDGES 160000

typedef __attribute__((ext_vector_type(16))) __bf16 v16bf;
typedef __attribute__((ext_vector_type(8)))  float  v8f;

// per-layer tensor-product weight counts and irreps dims
constexpr int NUMEL_[4] = {320, 416, 464, 496};
constexpr int HIN_D_[4] = {16, 28, 40, 44};   // dim of hin (in_tp irreps)
constexpr int OTP_D_[4] = {28, 40, 44, 44};   // dim of tp output (out_tp irreps)

// ---------------- tensor-product path tables ----------------
// type: 0=scalar*sh0->scalar, 1=scalar*sh1->vec, 2=vec*sh0->vec, 3=vec.sh1->scalar, 4=vec x sh1->vec
// coef = sqrt(2lo+1)/sqrt(fan) * w3j factor (1, 1/sqrt3 or 1/sqrt6)
struct TPPath { int type; int off; int m1; int xoff; int ooff; int mo; float coef; };

template<int L> struct TPTab;
template<> struct TPTab<0> {
  static constexpr int NP = 2;
  static constexpr TPPath P[NP] = {
    {0,   0, 16, 0,  0, 16, 0.25f},
    {1, 256, 16, 0, 16,  4, 0.25f},
  };
};
template<> struct TPTab<1> {
  static constexpr int NP = 5;
  static constexpr TPPath P[NP] = {
    {0,   0, 16,  0,  0, 16, 0.22360680f},
    {1, 256, 16,  0, 16,  4, 0.22360680f},
    {2, 320,  4, 16, 16,  4, 0.22360680f},
    {3, 336,  4, 16,  0, 16, 0.12909944f},
    {4, 400,  4, 16, 28,  4, 0.35355339f},
  };
};
template<> struct TPTab<2> {
  static constexpr int NP = 8;
  static constexpr TPPath P[NP] = {
    {0,   0, 16,  0,  0, 16, 0.22360680f},
    {1, 256, 16,  0, 16,  4, 0.20412415f},
    {2, 320,  4, 16, 16,  4, 0.20412415f},
    {3, 336,  4, 16,  0, 16, 0.12909944f},
    {4, 400,  4, 16, 28,  4, 0.25f},
    {2, 416,  4, 28, 28,  4, 0.35355339f},
    {4, 432,  4, 28, 16,  4, 0.14433757f},
    {3, 448,  4, 28, 40,  4, 0.28867513f},
  };
};
template<> struct TPTab<3> {
  static constexpr int NP = 10;
  static constexpr TPPath P[NP] = {
    {0,   0, 16,  0,  0, 16, 0.22360680f},
    {1, 256, 16,  0, 16,  4, 0.20412415f},
    {2, 320,  4, 16, 16,  4, 0.20412415f},
    {3, 336,  4, 16,  0, 16, 0.12909944f},
    {4, 400,  4, 16, 28,  4, 0.20412415f},
    {2, 416,  4, 28, 28,  4, 0.28867513f},
    {4, 432,  4, 28, 16,  4, 0.14433757f},
    {3, 448,  4, 28, 40,  4, 0.20412415f},
    {0, 464,  4, 40, 40,  4, 0.35355339f},
    {1, 480,  4, 40, 28,  4, 0.28867513f},
  };
};

// ---------------- small helper kernels ----------------
__global__ void __launch_bounds__(256)
k_deg(const int* __restrict__ eidx, float* __restrict__ deg)
{
  int e = blockIdx.x * 256 + threadIdx.x;
  if (e < N_EDGES) atomicAdd(&deg[eidx[N_EDGES + e]], 1.0f);
}

// f32 [K,N] -> bf16 transposed [N,K] (WMMA B fragments become contiguous)
__global__ void __launch_bounds__(256)
k_wt(const float* __restrict__ W, __bf16* __restrict__ Wt, int K, int N)
{
  int i = blockIdx.x * 256 + threadIdx.x;
  if (i >= K * N) return;
  int k = i / N, n = i - k * N;
  Wt[(long)n * K + k] = (__bf16)W[(long)k * N + n];
}

// ---------------- node embedding: layernorm + t-emb + node_mlp ----------------
__global__ void __launch_bounds__(32)
k_node_embed(const float* __restrict__ node_attr, const float* __restrict__ node_t,
             const float* __restrict__ g, const float* __restrict__ b,
             const float* __restrict__ W0, const float* __restrict__ b0,
             const float* __restrict__ W1, const float* __restrict__ b1,
             const float* __restrict__ W2, const float* __restrict__ b2,
             float* __restrict__ t_emb, float* __restrict__ h)
{
  const int node = blockIdx.x;
  const int lane = threadIdx.x;
  __shared__ float in_[288];
  __shared__ float s1[32];
  const float* x = node_attr + (long)node * 256;

  float s = 0.f;
  for (int i = lane; i < 256; i += 32) s += x[i];
  #pragma unroll
  for (int m = 16; m > 0; m >>= 1) s += __shfl_xor(s, m, 32);
  const float mu = s * (1.f / 256.f);
  float vs = 0.f;
  for (int i = lane; i < 256; i += 32) { float d = x[i] - mu; vs += d * d; }
  #pragma unroll
  for (int m = 16; m > 0; m >>= 1) vs += __shfl_xor(vs, m, 32);
  const float inv = rsqrtf(vs * (1.f / 256.f) + 1e-5f);
  for (int i = lane; i < 256; i += 32) in_[32 + i] = (x[i] - mu) * inv * g[i] + b[i];

  // sinusoidal(t_scaled, 32): ts = log(t/0.01)/log(1e4)*1e4
  const float t  = node_t[node];
  const float ts = (logf(t) + 4.605170185988091f) * 1085.736204893732f;
  const int   j  = lane & 15;
  const float fr = expf(-9.210340371976184f * (float)j * (1.f / 15.f));
  const float ang = ts * fr;
  const float te  = (lane < 16) ? sinf(ang) : cosf(ang);
  in_[lane] = te;
  t_emb[(long)node * 32 + lane] = te;
  __syncthreads();

  float v = b0[lane];
  for (int k = 0; k < 288; ++k) v += in_[k] * W0[k * 32 + lane];
  v = fmaxf(v, 0.f);
  s1[lane] = v; __syncthreads();
  float v2 = b1[lane];
  for (int k = 0; k < 32; ++k) v2 += s1[k] * W1[k * 32 + lane];
  v2 = fmaxf(v2, 0.f);
  __syncthreads(); s1[lane] = v2; __syncthreads();
  float v3 = b2[lane];
  for (int k = 0; k < 32; ++k) v3 += s1[k] * W2[k * 32 + lane];
  h[(long)node * 96 + lane] = v3;
}

// ---------------- edge embedding: layernorm + gauss + pos-emb + sh + edge_mlp ----------------
__global__ void __launch_bounds__(32)
k_edge_embed(const float* __restrict__ edge_attr, const int* __restrict__ eidx,
             const float* __restrict__ pos, const float* __restrict__ t_emb,
             const float* __restrict__ g, const float* __restrict__ b,
             const float* __restrict__ W0, const float* __restrict__ b0,
             const float* __restrict__ W1, const float* __restrict__ b1,
             const float* __restrict__ W2, const float* __restrict__ b2,
             float* __restrict__ e_feat, float* __restrict__ edge_sh)
{
  const int e = blockIdx.x;
  const int lane = threadIdx.x;
  const int src = eidx[e];
  const int dst = eidx[N_EDGES + e];
  __shared__ float in_[354];  // [gauss 0..49 | pe 50..65 | temb 66..97 | eln 98..353]
  __shared__ float s1[32];
  const float* x = edge_attr + (long)e * 256;

  float s = 0.f;
  for (int i = lane; i < 256; i += 32) s += x[i];
  #pragma unroll
  for (int m = 16; m > 0; m >>= 1) s += __shfl_xor(s, m, 32);
  const float mu = s * (1.f / 256.f);
  float vs = 0.f;
  for (int i = lane; i < 256; i += 32) { float d = x[i] - mu; vs += d * d; }
  #pragma unroll
  for (int m = 16; m > 0; m >>= 1) vs += __shfl_xor(vs, m, 32);
  const float inv = rsqrtf(vs * (1.f / 256.f) + 1e-5f);
  for (int i = lane; i < 256; i += 32) in_[98 + i] = (x[i] - mu) * inv * g[i] + b[i];

  const float dx = pos[src * 3 + 0] - pos[dst * 3 + 0];
  const float dy = pos[src * 3 + 1] - pos[dst * 3 + 1];
  const float dz = pos[src * 3 + 2] - pos[dst * 3 + 2];
  const float r  = sqrtf(dx * dx + dy * dy + dz * dz);

  const float step = 50.f / 49.f;
  const float coef = -0.5f / (step * step);
  for (int i = lane; i < 50; i += 32) { float d = r - step * (float)i; in_[i] = expf(coef * d * d); }

  const float xd = (float)(dst - src);
  if (lane < 16) {
    const int   jj  = lane & 7;
    const float fr  = expf(-9.210340371976184f * (float)jj * (1.f / 7.f));
    const float ang = xd * fr;
    in_[50 + lane] = (lane < 8) ? sinf(ang) : cosf(ang);
  }
  in_[66 + lane] = t_emb[(long)src * 32 + lane];

  if (lane == 0) {
    const float nr = fmaxf(r, 1e-9f);
    const float s3 = 1.7320508075688772f;
    edge_sh[(long)e * 4 + 0] = 1.f;
    edge_sh[(long)e * 4 + 1] = s3 * dy / nr;
    edge_sh[(long)e * 4 + 2] = s3 * dz / nr;
    edge_sh[(long)e * 4 + 3] = s3 * dx / nr;
  }
  __syncthreads();

  float v = b0[lane];
  for (int k = 0; k < 354; ++k) v += in_[k] * W0[k * 32 + lane];
  v = fmaxf(v, 0.f);
  s1[lane] = v; __syncthreads();
  float v2 = b1[lane];
  for (int k = 0; k < 32; ++k) v2 += s1[k] * W1[k * 32 + lane];
  v2 = fmaxf(v2, 0.f);
  __syncthreads(); s1[lane] = v2; __syncthreads();
  float v3 = b2[lane];
  for (int k = 0; k < 32; ++k) v3 += s1[k] * W2[k * 32 + lane];
  e_feat[(long)e * 32 + lane] = v3;
}

// ---------------- o3 linear into TP space (per node) ----------------
template<int L>
__global__ void __launch_bounds__(256)
k_lin_in(const float* __restrict__ h,
         const float* __restrict__ W00, const float* __restrict__ W11,
         const float* __restrict__ W22, const float* __restrict__ W33,
         float* __restrict__ hin)
{
  const int n = blockIdx.x * 256 + threadIdx.x;
  if (n >= N_NODES) return;
  const float* hr = h + (long)n * 96;
  float* o = hin + (long)n * 48;
  const float c32 = 0.17677669529663687f;  // 1/sqrt(32)
  const float c8  = 0.35355339059327373f;  // 1/sqrt(8)
  for (int v = 0; v < 16; ++v) {
    float a = 0.f;
    for (int u = 0; u < 32; ++u) a += hr[u] * W00[u * 16 + v];
    o[v] = a * c32;
  }
  if constexpr (L >= 1) {
    for (int v = 0; v < 4; ++v)
      for (int k = 0; k < 3; ++k) {
        float a = 0.f;
        for (int u = 0; u < 8; ++u) a += hr[32 + u * 3 + k] * W11[u * 4 + v];
        o[16 + v * 3 + k] = a * c8;
      }
  }
  if constexpr (L >= 2) {
    for (int v = 0; v < 4; ++v)
      for (int k = 0; k < 3; ++k) {
        float a = 0.f;
        for (int u = 0; u < 8; ++u) a += hr[56 + u * 3 + k] * W22[u * 4 + v];
        o[28 + v * 3 + k] = a * c8;
      }
  }
  if constexpr (L >= 3) {
    for (int v = 0; v < 4; ++v) {
      float a = 0.f;
      for (int u = 0; u < 8; ++u) a += hr[80 + u] * W33[u * 4 + v];
      o[40 + v] = a * c8;
    }
  }
}

// ---------------- o3 linear out of TP space + residual-pad (per node) ----------------
template<int L>
__global__ void __launch_bounds__(256)
k_lin_out(const float* __restrict__ agg, const float* __restrict__ deg,
          const float* __restrict__ W00, const float* __restrict__ W11,
          const float* __restrict__ W22, const float* __restrict__ W33,
          const float* __restrict__ hp, float* __restrict__ hn)
{
  const int n = blockIdx.x * 256 + threadIdx.x;
  if (n >= N_NODES) return;
  const float inv = 1.f / fmaxf(deg[n], 1.f);
  const float* a  = agg + (long)n * 48;
  const float* h0 = hp + (long)n * 96;
  float* o = hn + (long)n * 96;
  for (int v = 0; v < 32; ++v) {                        // 16x0e -> 32x0e
    float x = 0.f;
    for (int u = 0; u < 16; ++u) x += a[u] * W00[u * 32 + v];
    o[v] = 0.25f * inv * x + h0[v];
  }
  for (int v = 0; v < 8; ++v)                           // 4x1o -> 8x1o
    for (int k = 0; k < 3; ++k) {
      float x = 0.f;
      for (int u = 0; u < 4; ++u) x += a[16 + u * 3 + k] * W11[u * 8 + v];
      float rv = 0.5f * inv * x;
      if constexpr (L >= 1) rv += h0[32 + v * 3 + k];
      o[32 + v * 3 + k] = rv;
    }
  if constexpr (L >= 1) {                               // 4x1e -> 8x1e
    for (int v = 0; v < 8; ++v)
      for (int k = 0; k < 3; ++k) {
        float x = 0.f;
        for (int u = 0; u < 4; ++u) x += a[28 + u * 3 + k] * W22[u * 8 + v];
        float rv = 0.5f * inv * x;
        if constexpr (L >= 2) rv += h0[56 + v * 3 + k];
        o[56 + v * 3 + k] = rv;
      }
  }
  if constexpr (L >= 2) {                               // 4x0o -> 8x0o
    for (int v = 0; v < 8; ++v) {
      float x = 0.f;
      for (int u = 0; u < 4; ++u) x += a[40 + u] * W33[u * 8 + v];
      float rv = 0.5f * inv * x;
      if constexpr (L >= 3) rv += h0[80 + v];
      o[80 + v] = rv;
    }
  }
}

// ---------------- THE HOT KERNEL ----------------
// 32 edges per workgroup (M=32): each wave loads one B fragment and feeds TWO
// v_wmma_f32_16x16x32_bf16 (two 16-row A tiles), halving global B traffic so
// the XDL pipe is fed at half the vector-load bandwidth per MAC.
template<int L>
__global__ void __launch_bounds__(256)
k_edge_fused(const int* __restrict__ eidx,
             const float* __restrict__ e_feat,
             const float* __restrict__ h,
             const float* __restrict__ hin,
             const float* __restrict__ edge_sh,
             const __bf16* __restrict__ wt1,   // [128 x 96]  transposed bf16
             const __bf16* __restrict__ wt2,   // [128 x 128]
             const __bf16* __restrict__ wt3,   // [NUMEL x 128]
             const float* __restrict__ fb0,
             const float* __restrict__ fb1,
             const float* __restrict__ fb2,
             float* __restrict__ agg)
{
  constexpr int NUMEL = NUMEL_[L];
  constexpr int NT3   = NUMEL / 16;
  constexpr int HD    = HIN_D_[L];
  constexpr int OD    = OTP_D_[L];

  __shared__ __bf16 eaS[32 * 104];    // ea: 32 edges x 96 (+pad)
  __shared__ __bf16 a1S[32 * 136];    // act1: 32 x 128 (+pad)
  __shared__ __bf16 a2S[32 * 136];    // act2
  __shared__ float  wS[32 * NUMEL];   // per-edge TP weights
  __shared__ float  hinS[32 * 48];
  __shared__ float  shS[32 * 4];
  __shared__ int    srcS[32], dstS[32];

  const int tid  = threadIdx.x;
  const int lane = tid & 31;
  const int wv   = tid >> 5;              // wave 0..7
  const int hi   = (lane >> 4) & 1;
  const int colL = lane & 15;
  const long e0  = (long)blockIdx.x * 32;

  if (tid < 32) {
    srcS[tid] = eidx[e0 + tid];
    dstS[tid] = eidx[N_EDGES + e0 + tid];
  }
  __syncthreads();

  // stage ea = [e_feat | h[src][:32] | h[dst][:32]] as bf16 (32x96 = 12*256)
  #pragma unroll
  for (int i = 0; i < 12; ++i) {
    int idx = tid + i * 256;
    int row = idx / 96, col = idx - row * 96;
    float v;
    if (col < 32)      v = e_feat[(e0 + row) * 32 + col];
    else if (col < 64) v = h[(long)srcS[row] * 96 + (col - 32)];
    else               v = h[(long)dstS[row] * 96 + (col - 64)];
    eaS[row * 104 + col] = (__bf16)v;
  }
  for (int i = tid; i < 32 * HD; i += 256) {
    int row = i / HD, c = i - row * HD;
    hinS[row * 48 + c] = hin[(long)srcS[row] * 48 + c];
  }
  if (tid < 128) shS[tid] = edge_sh[(e0 + (tid >> 2)) * 4 + (tid & 3)];
  __syncthreads();

  const int rowA = lane & 15;
  const int kbA  = hi ? 8 : 0;    // A-frag K-offset (16-bit A 16x32 layout)
  const int kbB  = hi ? 8 : 0;    // B-frag K-offset (same K order as A-frag)
  const int rofs = hi ? 8 : 0;    // D-frag row offset

  // ---- fc1: [32x96] @ [96x128], relu ----
  {
    v8f ac0 = {0.f,0.f,0.f,0.f,0.f,0.f,0.f,0.f};
    v8f ac1 = {0.f,0.f,0.f,0.f,0.f,0.f,0.f,0.f};
    #pragma unroll
    for (int kf = 0; kf < 3; ++kf) {
      const int k0 = kf * 32;
      v16bf A0, A1, B;
      const __bf16* pb = &wt1[(long)(wv * 16 + colL) * 96 + k0 + kbB];
      #pragma unroll
      for (int q = 0; q < 8; ++q) { B[q] = pb[q]; B[8 + q] = pb[16 + q]; }
      const __bf16* pa0 = &eaS[rowA * 104 + k0 + kbA];
      const __bf16* pa1 = &eaS[(16 + rowA) * 104 + k0 + kbA];
      #pragma unroll
      for (int q = 0; q < 8; ++q) { A0[q] = pa0[q]; A0[8 + q] = pa0[16 + q];
                                    A1[q] = pa1[q]; A1[8 + q] = pa1[16 + q]; }
      ac0 = __builtin_amdgcn_wmma_f32_16x16x32_bf16(false, A0, false, B, (short)0, ac0, false, false);
      ac1 = __builtin_amdgcn_wmma_f32_16x16x32_bf16(false, A1, false, B, (short)0, ac1, false, false);
    }
    const int col = wv * 16 + colL;
    const float bias = fb0[col];
    #pragma unroll
    for (int r = 0; r < 8; ++r) {
      a1S[(r + rofs)      * 136 + col] = (__bf16)fmaxf(ac0[r] + bias, 0.f);
      a1S[(16 + r + rofs) * 136 + col] = (__bf16)fmaxf(ac1[r] + bias, 0.f);
    }
  }
  __syncthreads();

  // ---- fc2: [32x128] @ [128x128], relu ----
  {
    v8f ac0 = {0.f,0.f,0.f,0.f,0.f,0.f,0.f,0.f};
    v8f ac1 = {0.f,0.f,0.f,0.f,0.f,0.f,0.f,0.f};
    #pragma unroll
    for (int kf = 0; kf < 4; ++kf) {
      const int k0 = kf * 32;
      v16bf A0, A1, B;
      const __bf16* pb = &wt2[(long)(wv * 16 + colL) * 128 + k0 + kbB];
      #pragma unroll
      for (int q = 0; q < 8; ++q) { B[q] = pb[q]; B[8 + q] = pb[16 + q]; }
      const __bf16* pa0 = &a1S[rowA * 136 + k0 + kbA];
      const __bf16* pa1 = &a1S[(16 + rowA) * 136 + k0 + kbA];
      #pragma unroll
      for (int q = 0; q < 8; ++q) { A0[q] = pa0[q]; A0[8 + q] = pa0[16 + q];
                                    A1[q] = pa1[q]; A1[8 + q] = pa1[16 + q]; }
      ac0 = __builtin_amdgcn_wmma_f32_16x16x32_bf16(false, A0, false, B, (short)0, ac0, false, false);
      ac1 = __builtin_amdgcn_wmma_f32_16x16x32_bf16(false, A1, false, B, (short)0, ac1, false, false);
    }
    const int col = wv * 16 + colL;
    const float bias = fb1[col];
    #pragma unroll
    for (int r = 0; r < 8; ++r) {
      a2S[(r + rofs)      * 136 + col] = (__bf16)fmaxf(ac0[r] + bias, 0.f);
      a2S[(16 + r + rofs) * 136 + col] = (__bf16)fmaxf(ac1[r] + bias, 0.f);
    }
  }
  __syncthreads();

  // ---- fc3: [32x128] @ [128xNUMEL], linear -> f32 in LDS ----
  for (int tile = wv; tile < NT3; tile += 8) {
    v8f ac0 = {0.f,0.f,0.f,0.f,0.f,0.f,0.f,0.f};
    v8f ac1 = {0.f,0.f,0.f,0.f,0.f,0.f,0.f,0.f};
    #pragma unroll
    for (int kf = 0; kf < 4; ++kf) {
      const int k0 = kf * 32;
      v16bf A0, A1, B;
      const __bf16* pb = &wt3[(long)(tile * 16 + colL) * 128 + k0 + kbB];
      #pragma unroll
      for (int q = 0; q < 8; ++q) { B[q] = pb[q]; B[8 + q] = pb[16 + q]; }
      const __bf16* pa0 = &a2S[rowA * 136 + k0 + kbA];
      const __bf16* pa1 = &a2S[(16 + rowA) * 136 + k0 + kbA];
      #pragma unroll
      for (int q = 0; q < 8; ++q) { A0[q] = pa0[q]; A0[8 + q] = pa0[16 + q];
                                    A1[q] = pa1[q]; A1[8 + q] = pa1[16 + q]; }
      ac0 = __builtin_amdgcn_wmma_f32_16x16x32_bf16(false, A0, false, B, (short)0, ac0, false, false);
      ac1 = __builtin_amdgcn_wmma_f32_16x16x32_bf16(false, A1, false, B, (short)0, ac1, false, false);
    }
    const float bias = fb2[tile * 16 + colL];
    #pragma unroll
    for (int r = 0; r < 8; ++r) {
      wS[(r + rofs)      * NUMEL + tile * 16 + colL] = ac0[r] + bias;
      wS[(16 + r + rofs) * NUMEL + tile * 16 + colL] = ac1[r] + bias;
    }
  }
  __syncthreads();

  // ---- tensor product + scatter-add (16 threads per edge, two passes) ----
  #pragma unroll
  for (int eg = 0; eg < 2; ++eg) {
    const int et = eg * 16 + (tid >> 4);
    const int t  = tid & 15;
    const float* xv = &hinS[et * 48];
    const float sa = shS[et * 4 + 1], sb = shS[et * 4 + 2], sc = shS[et * 4 + 3];
    const float* wp = &wS[et * NUMEL];
    const int w3 = t / 3, k3 = t - 3 * w3;
    float acc0 = 0.f, acc1 = 0.f, acc2 = 0.f, acc3 = 0.f;
    #pragma unroll
    for (int pi = 0; pi < TPTab<L>::NP; ++pi) {
      const TPPath pp = TPTab<L>::P[pi];
      if (pp.type == 0) {                       // scalar * sh0 -> scalar
        if (t < pp.mo) {
          float s = 0.f;
          for (int u = 0; u < pp.m1; ++u) s += xv[pp.xoff + u] * wp[pp.off + u * pp.mo + t];
          if (pp.ooff == 0) acc0 += pp.coef * s; else acc3 += pp.coef * s;
        }
      } else if (pp.type == 1) {                // scalar * sh1 -> vector
        if (t < 3 * pp.mo) {
          float s = 0.f;
          for (int u = 0; u < pp.m1; ++u) s += xv[pp.xoff + u] * wp[pp.off + u * pp.mo + w3];
          const float shk = (k3 == 0) ? sa : (k3 == 1 ? sb : sc);
          if (pp.ooff == 16) acc1 += pp.coef * s * shk; else acc2 += pp.coef * s * shk;
        }
      } else if (pp.type == 2) {                // vector * sh0 -> vector
        if (t < 3 * pp.mo) {
          float s = 0.f;
          for (int u = 0; u < pp.m1; ++u) s += xv[pp.xoff + u * 3 + k3] * wp[pp.off + u * pp.mo + w3];
          if (pp.ooff == 16) acc1 += pp.coef * s; else acc2 += pp.coef * s;
        }
      } else if (pp.type == 3) {                // vector . sh1 -> scalar
        if (t < pp.mo) {
          float s = 0.f;
          for (int u = 0; u < pp.m1; ++u) {
            const float d = xv[pp.xoff + u*3] * sa + xv[pp.xoff + u*3 + 1] * sb + xv[pp.xoff + u*3 + 2] * sc;
            s += d * wp[pp.off + u * pp.mo + t];
          }
          if (pp.ooff == 0) acc0 += pp.coef * s; else acc3 += pp.coef * s;
        }
      } else {                                  // vector x sh1 -> vector
        if (t < 3 * pp.mo) {
          float s = 0.f;
          for (int u = 0; u < pp.m1; ++u) {
            const float x0 = xv[pp.xoff + u*3], x1 = xv[pp.xoff + u*3 + 1], x2 = xv[pp.xoff + u*3 + 2];
            float cr;
            if (k3 == 0)      cr = x1 * sc - x2 * sb;
            else if (k3 == 1) cr = x2 * sa - x0 * sc;
            else              cr = x0 * sb - x1 * sa;
            s += cr * wp[pp.off + u * pp.mo + w3];
          }
          if (pp.ooff == 16) acc1 += pp.coef * s; else acc2 += pp.coef * s;
        }
      }
    }
    float* ag = agg + (long)dstS[et] * 48;
    atomicAdd(&ag[t], acc0);
    if (OD > 16 && t < 12) atomicAdd(&ag[16 + t], acc1);
    if (OD > 28 && t < 12) atomicAdd(&ag[28 + t], acc2);
    if (OD > 40 && t < 4)  atomicAdd(&ag[40 + t], acc3);
  }
}

// ---------------- final TP: fctp(h, h, final_tp) -> 1x1o, * score_norm ----------------
__global__ void __launch_bounds__(256)
k_final(const float* __restrict__ h, const float* __restrict__ w,
        const float* __restrict__ sn, float* __restrict__ out)
{
  const int n = blockIdx.x * 256 + threadIdx.x;
  if (n >= N_NODES) return;
  const float* x   = h + (long)n * 96;
  const float* s   = x;        // 32x0e
  const float* v1o = x + 32;   // 8x1o
  const float* v1e = x + 56;   // 8x1e
  const float* s0o = x + 80;   // 8x0o
  float o0 = 0.f, o1 = 0.f, o2 = 0.f;
  const float cA = 0.03608439182435161f;  // (sqrt3/sqrt768)/sqrt3 = 1/16/sqrt3
  const float cB = 0.02551551815399144f;  // 1/16/sqrt6
  // P1: 0e(32) x 1o(8), off 0
  for (int v = 0; v < 8; ++v) {
    float ts = 0.f;
    for (int u = 0; u < 32; ++u) ts += s[u] * w[u * 8 + v];
    o0 += cA * ts * v1o[v*3+0]; o1 += cA * ts * v1o[v*3+1]; o2 += cA * ts * v1o[v*3+2];
  }
  // P2: 1o(8) x 0e(32), off 256
  for (int u = 0; u < 8; ++u) {
    float ts = 0.f;
    for (int v = 0; v < 32; ++v) ts += s[v] * w[256 + u * 32 + v];
    o0 += cA * ts * v1o[u*3+0]; o1 += cA * ts * v1o[u*3+1]; o2 += cA * ts * v1o[u*3+2];
  }
  // P3: 1o(8) x 1e(8) cross, off 512
  for (int u = 0; u < 8; ++u) {
    float m0 = 0.f, m1 = 0.f, m2 = 0.f;
    for (int v = 0; v < 8; ++v) {
      const float ww = w[512 + u * 8 + v];
      m0 += v1e[v*3+0] * ww; m1 += v1e[v*3+1] * ww; m2 += v1e[v*3+2] * ww;
    }
    const float a0 = v1o[u*3+0], a1 = v1o[u*3+1], a2 = v1o[u*3+2];
    o0 += cB * (a1 * m2 - a2 * m1); o1 += cB * (a2 * m0 - a0 * m2); o2 += cB * (a0 * m1 - a1 * m0);
  }
  // P4: 1e(8) x 1o(8) cross, off 576
  for (int u = 0; u < 8; ++u) {
    float m0 = 0.f, m1 = 0.f, m2 = 0.f;
    for (int v = 0; v < 8; ++v) {
      const float ww = w[576 + u * 8 + v];
      m0 += v1o[v*3+0] * ww; m1 += v1o[v*3+1] * ww; m2 += v1o[v*3+2] * ww;
    }
    const float a0 = v1e[u*3+0], a1 = v1e[u*3+1], a2 = v1e[u*3+2];
    o0 += cB * (a1 * m2 - a2 * m1); o1 += cB * (a2 * m0 - a0 * m2); o2 += cB * (a0 * m1 - a1 * m0);
  }
  // P5: 1e(8) x 0o(8), off 640
  for (int u = 0; u < 8; ++u) {
    float ts = 0.f;
    for (int v = 0; v < 8; ++v) ts += s0o[v] * w[640 + u * 8 + v];
    o0 += cA * ts * v1e[u*3+0]; o1 += cA * ts * v1e[u*3+1]; o2 += cA * ts * v1e[u*3+2];
  }
  // P6: 0o(8) x 1e(8), off 704
  for (int v = 0; v < 8; ++v) {
    float ts = 0.f;
    for (int u = 0; u < 8; ++u) ts += s0o[u] * w[704 + u * 8 + v];
    o0 += cA * ts * v1e[v*3+0]; o1 += cA * ts * v1e[v*3+1]; o2 += cA * ts * v1e[v*3+2];
  }
  const float q = sn[n];
  out[n*3+0] = o0 * q; out[n*3+1] = o1 * q; out[n*3+2] = o2 * q;
}

// ---------------- driver ----------------
extern "C" void kernel_launch(void* const* d_in, const int* in_sizes, int n_in,
                              void* d_out, int out_size, void* d_ws, size_t ws_size,
                              hipStream_t stream)
{
  (void)in_sizes; (void)n_in; (void)out_size; (void)ws_size;
  auto F = [&](int i) { return (const float*)d_in[i]; };
  const int* EIX = (const int*)d_in[4];

  char* wptr = (char*)d_ws;
  auto alloc = [&](size_t bytes) { char* p = wptr; wptr += (bytes + 255) & ~(size_t)255; return p; };
  float*  t_emb  = (float*)alloc((size_t)N_NODES * 32 * 4);
  float*  hA     = (float*)alloc((size_t)N_NODES * 96 * 4);
  float*  hB     = (float*)alloc((size_t)N_NODES * 96 * 4);
  float*  hin    = (float*)alloc((size_t)N_NODES * 48 * 4);
  float*  agg    = (float*)alloc((size_t)N_NODES * 48 * 4);
  float*  deg    = (float*)alloc((size_t)N_NODES * 4);
  float*  e_feat = (float*)alloc((size_t)N_EDGES * 32 * 4);
  float*  esh    = (float*)alloc((size_t)N_EDGES * 4 * 4);
  __bf16* wtb    = (__bf16*)alloc((size_t)331776 * 2);

  size_t wto[4];
  { size_t a = 0; for (int l = 0; l < 4; ++l) { wto[l] = a; a += 12288 + 16384 + (size_t)NUMEL_[l] * 128; } }
  const int LB[4] = {22, 31, 42, 55};  // param-leaf base index per layer (insertion order)

  hipMemsetAsync(deg, 0, (size_t)N_NODES * 4, stream);
  k_deg<<<(N_EDGES + 255) / 256, 256, 0, stream>>>(EIX, deg);

  for (int l = 0; l < 4; ++l) {
    const int fcb = LB[l] + (l + 1);
    k_wt<<<(96 * 128 + 255) / 256, 256, 0, stream>>>(F(fcb),     wtb + wto[l],          96, 128);
    k_wt<<<(128 * 128 + 255) / 256, 256, 0, stream>>>(F(fcb + 2), wtb + wto[l] + 12288, 128, 128);
    k_wt<<<(128 * NUMEL_[l] + 255) / 256, 256, 0, stream>>>(F(fcb + 4), wtb + wto[l] + 28672, 128, NUMEL_[l]);
  }

  k_node_embed<<<N_NODES, 32, 0, stream>>>(F(0), F(2), F(6), F(7),
      F(10), F(11), F(12), F(13), F(14), F(15), t_emb, hA);
  k_edge_embed<<<N_EDGES, 32, 0, stream>>>(F(1), EIX, F(3), t_emb, F(8), F(9),
      F(16), F(17), F(18), F(19), F(20), F(21), e_feat, esh);

  const int NB = (N_NODES + 255) / 256;
  const int EB = N_EDGES / 32;

  // layer 0: hA -> hB
  k_lin_in<0><<<NB, 256, 0, stream>>>(hA, F(22), nullptr, nullptr, nullptr, hin);
  hipMemsetAsync(agg, 0, (size_t)N_NODES * 48 * 4, stream);
  k_edge_fused<0><<<EB, 256, 0, stream>>>(EIX, e_feat, hA, hin, esh,
      wtb + wto[0], wtb + wto[0] + 12288, wtb + wto[0] + 28672, F(24), F(26), F(28), agg);
  k_lin_out<0><<<NB, 256, 0, stream>>>(agg, deg, F(29), F(30), nullptr, nullptr, hA, hB);

  // layer 1: hB -> hA
  k_lin_in<1><<<NB, 256, 0, stream>>>(hB, F(31), F(32), nullptr, nullptr, hin);
  hipMemsetAsync(agg, 0, (size_t)N_NODES * 48 * 4, stream);
  k_edge_fused<1><<<EB, 256, 0, stream>>>(EIX, e_feat, hB, hin, esh,
      wtb + wto[1], wtb + wto[1] + 12288, wtb + wto[1] + 28672, F(34), F(36), F(38), agg);
  k_lin_out<1><<<NB, 256, 0, stream>>>(agg, deg, F(39), F(40), F(41), nullptr, hB, hA);

  // layer 2: hA -> hB
  k_lin_in<2><<<NB, 256, 0, stream>>>(hA, F(42), F(43), F(44), nullptr, hin);
  hipMemsetAsync(agg, 0, (size_t)N_NODES * 48 * 4, stream);
  k_edge_fused<2><<<EB, 256, 0, stream>>>(EIX, e_feat, hA, hin, esh,
      wtb + wto[2], wtb + wto[2] + 12288, wtb + wto[2] + 28672, F(46), F(48), F(50), agg);
  k_lin_out<2><<<NB, 256, 0, stream>>>(agg, deg, F(51), F(52), F(53), F(54), hA, hB);

  // layer 3: hB -> hA
  k_lin_in<3><<<NB, 256, 0, stream>>>(hB, F(55), F(56), F(57), F(58), hin);
  hipMemsetAsync(agg, 0, (size_t)N_NODES * 48 * 4, stream);
  k_edge_fused<3><<<EB, 256, 0, stream>>>(EIX, e_feat, hB, hin, esh,
      wtb + wto[3], wtb + wto[3] + 12288, wtb + wto[3] + 28672, F(60), F(62), F(64), agg);
  k_lin_out<3><<<NB, 256, 0, stream>>>(agg, deg, F(65), F(66), F(67), F(68), hB, hA);

  k_final<<<NB, 256, 0, stream>>>(hA, F(69), F(5), (float*)d_out);
}
